// QBlock_23605140259270
// MI455X (gfx1250) — 8-bit matrix operand order repaired and checked by reading, compiled, not run on hardware
//
#include <hip/hip_runtime.h>
#include <hip/hip_bf16.h>


#define Bn   8
#define Cc   128
#define Hh   192
#define Ww   192
#define HWp  (Hh * Ww)
#define CHW  (Cc * HWp)
#define TOT  (Bn * CHW)

typedef __attribute__((ext_vector_type(8))) int       v8i;
typedef __attribute__((ext_vector_type(8))) _Float16  h8;

__device__ inline int qclamp8(float v, float inv_s) {
    int q = __float2int_rn(v * inv_s);
    return min(127, max(-128, q));
}

__device__ inline void load8f(const float* p, float v[8]) {
    const float4 a = *(const float4*)p;
    const float4 b = *(const float4*)(p + 4);
    v[0] = a.x; v[1] = a.y; v[2] = a.z; v[3] = a.w;
    v[4] = b.x; v[5] = b.y; v[6] = b.z; v[7] = b.w;
}
__device__ inline void load8f(const _Float16* p, float v[8]) {
    const h8 hv = *(const h8*)p;
    #pragma unroll
    for (int k = 0; k < 8; ++k) v[k] = (float)hv[k];
}

__global__ void k_init(unsigned* sc) {
    if (threadIdx.x < 16) sc[threadIdx.x] = 0u;
}

__global__ void k_absmax_f32(const float* __restrict__ x, long long n,
                             unsigned* __restrict__ slot) {
    __shared__ unsigned sm;
    if (threadIdx.x == 0) sm = 0u;
    __syncthreads();
    float m = 0.0f;
    for (long long i = blockIdx.x * (long long)blockDim.x + threadIdx.x;
         i < n; i += (long long)gridDim.x * blockDim.x)
        m = fmaxf(m, fabsf(x[i]));
    atomicMax(&sm, __float_as_uint(m));
    __syncthreads();
    if (threadIdx.x == 0) atomicMax(slot, sm);
}

__global__ void k_prep(const float* __restrict__ w1, const float* __restrict__ p1,
                       const float* __restrict__ w2, const float* __restrict__ p2,
                       float* __restrict__ wscale,
                       signed char* __restrict__ w1q, signed char* __restrict__ p1q,
                       signed char* __restrict__ w2q, signed char* __restrict__ p2q) {
    __shared__ unsigned sm;
    const int tid = threadIdx.x;
    const float* src[4] = { w1, p1, w2, p2 };
    signed char* dst[4] = { w1q, p1q, w2q, p2q };
    const int    nel[4] = { Cc * 9, Cc * Cc, Cc * 9, Cc * Cc };
    for (int t = 0; t < 4; ++t) {
        if (tid == 0) sm = 0u;
        __syncthreads();
        float m = 0.0f;
        for (int i = tid; i < nel[t]; i += blockDim.x)
            m = fmaxf(m, fabsf(src[t][i]));
        atomicMax(&sm, __float_as_uint(m));
        __syncthreads();
        const float s   = fmaxf(__uint_as_float(sm) / 7.0f, 1e-8f);
        const float inv = 1.0f / s;
        if (tid == 0) wscale[t] = s;
        for (int i = tid; i < nel[t]; i += blockDim.x) {
            int q = __float2int_rn(src[t][i] * inv);
            q = min(7, max(-8, q));
            dst[t][i] = (signed char)q;
        }
        __syncthreads();
    }
}

template <typename TIN>
__global__ void k_dw(const TIN* __restrict__ in, const signed char* __restrict__ wq,
                     const float* __restrict__ bias,
                     const unsigned* __restrict__ actSlot,
                     const float* __restrict__ wScale,
                     _Float16* __restrict__ out, unsigned* __restrict__ outSlot) {
    __shared__ unsigned sm;
    if (threadIdx.x == 0) sm = 0u;
    __syncthreads();

    const long long idx0 =
        (blockIdx.x * (long long)blockDim.x + threadIdx.x) * 8;
    float lmax = 0.0f;
    if (idx0 < (long long)TOT) {
        const int w0 = (int)(idx0 % Ww);
        long long t  = idx0 / Ww;
        const int h  = (int)(t % Hh); t /= Hh;
        const int c  = (int)(t % Cc);
        const int b  = (int)(t / Cc);

        const float sa     = fmaxf(__uint_as_float(*actSlot) / 127.0f, 1e-8f);
        const float inv_sa = 1.0f / sa;

        const TIN* base = in + (long long)(b * Cc + c) * HWp;
        const signed char* wk = wq + c * 9;
        int wkr[9];
        #pragma unroll
        for (int i = 0; i < 9; ++i) wkr[i] = (int)wk[i];

        int acc[8] = { 0, 0, 0, 0, 0, 0, 0, 0 };
        #pragma unroll
        for (int dy = -1; dy <= 1; ++dy) {
            const int hh = h + dy;
            if ((unsigned)hh < (unsigned)Hh) {
                const TIN* row = base + hh * Ww + w0;
                float v[8];
                load8f(row, v);
                int q[10];
                q[0] = (w0 > 0)      ? qclamp8((float)row[-1], inv_sa) : 0;
                q[9] = (w0 + 8 < Ww) ? qclamp8((float)row[8],  inv_sa) : 0;
                #pragma unroll
                for (int k = 0; k < 8; ++k) q[k + 1] = qclamp8(v[k], inv_sa);
                const int wr0 = wkr[(dy + 1) * 3 + 0];
                const int wr1 = wkr[(dy + 1) * 3 + 1];
                const int wr2 = wkr[(dy + 1) * 3 + 2];
                #pragma unroll
                for (int k = 0; k < 8; ++k)
                    acc[k] += q[k] * wr0 + q[k + 1] * wr1 + q[k + 2] * wr2;
            }
        }
        const float dq = sa * wScale[0];
        const float bc = bias[c];
        h8 ov;
        #pragma unroll
        for (int k = 0; k < 8; ++k) {
            const float y = (float)acc[k] * dq + bc;
            ov[k] = (_Float16)y;
            lmax = fmaxf(lmax, fabsf(y));
        }
        *(h8*)(out + idx0) = ov;
    }
    atomicMax(&sm, __float_as_uint(lmax));
    __syncthreads();
    if (threadIdx.x == 0) atomicMax(outSlot, sm);
}

template <bool RELU, bool RESID, typename TOUT>
__global__ void k_pw(const _Float16* __restrict__ act, const signed char* __restrict__ wq,
                     const float* __restrict__ bias,
                     const unsigned* __restrict__ actSlot,
                     const float* __restrict__ wScalePtr,
                     TOUT* __restrict__ out, const float* __restrict__ resid,
                     unsigned* __restrict__ outSlot) {
    __shared__ unsigned char actB[64 * 128];
    __shared__ unsigned      wS[128 * 32];
    __shared__ float         biasS[128];
    __shared__ unsigned      sm;

    const int tid     = threadIdx.x;
    const int blk     = blockIdx.x;
    const int b       = blk / (HWp / 64);
    const int pixBase = (blk % (HWp / 64)) * 64;

    if (tid == 0) sm = 0u;

    const float sa     = fmaxf(__uint_as_float(*actSlot) / 127.0f, 1e-8f);
    const float inv_sa = 1.0f / sa;
    const float deq    = sa * wScalePtr[0];

    if (tid < 128) biasS[tid] = bias[tid];

    {
        const _Float16* actImg = act + (long long)b * CHW + pixBase;
        unsigned* actD = (unsigned*)actB;
        const int cin4 = tid >> 3;
        const int p0   = (tid & 7) * 8;
        int q[4][8];
        #pragma unroll
        for (int r = 0; r < 4; ++r) {
            const h8 hv = *(const h8*)(actImg + (long long)(cin4 * 4 + r) * HWp + p0);
            #pragma unroll
            for (int k = 0; k < 8; ++k)
                q[r][k] = qclamp8((float)hv[k], inv_sa);
        }
        #pragma unroll
        for (int k = 0; k < 8; ++k) {
            const unsigned d = ((unsigned)(q[0][k] & 255))
                             | ((unsigned)(q[1][k] & 255) << 8)
                             | ((unsigned)(q[2][k] & 255) << 16)
                             | ((unsigned)(q[3][k] & 255) << 24);
            actD[(p0 + k) * 32 + cin4] = d;
        }
    }
    {
        __builtin_prefetch(wq, 0, 3);
        const uint4* wsrc = (const uint4*)wq;
        uint4*       wdst = (uint4*)wS;
        for (int i = tid; i < 1024; i += 256) wdst[i] = wsrc[i];
    }
    __syncthreads();

    const int lane  = tid & 31;
    const int wave  = tid >> 5;
    const int l16   = lane >> 4;
    const int lm    = lane & 15;
    const int coutT = wave * 16;

    v8i a0, a1;
    {
        const unsigned* wrow = &wS[(coutT + lm) * 32];
        const int off = l16 * 4;
        #pragma unroll
        for (int i = 0; i < 8; ++i) {
            const int pat = ((i >> 2) << 3) + (i & 3);
            a0[i] = (int)wrow[pat + off];
            a1[i] = (int)wrow[16 + pat + off];
        }
    }

    for (int pp = 0; pp < 2; ++pp) {
        const int ptx = pp * 2, pty = pp * 2 + 1;
        const unsigned* arx = (const unsigned*)&actB[(ptx * 16 + lm) * 128];
        const unsigned* ary = (const unsigned*)&actB[(pty * 16 + lm) * 128];
        v8i b0x, b1x, b0y, b1y;
        #pragma unroll
        for (int i = 0; i < 4; ++i) {
            b0x[i]     = (int)arx[i + l16 * 4];
            b0x[i + 4] = (int)arx[8 + i + l16 * 4];
            b1x[i]     = (int)arx[16 + i + l16 * 4];
            b1x[i + 4] = (int)arx[16 + 8 + i + l16 * 4];
            b0y[i]     = (int)ary[i + l16 * 4];
            b0y[i + 4] = (int)ary[8 + i + l16 * 4];
            b1y[i]     = (int)ary[16 + i + l16 * 4];
            b1y[i + 4] = (int)ary[16 + 8 + i + l16 * 4];
        }
        const v8i zc = { 0, 0, 0, 0, 0, 0, 0, 0 };
        v8i cx = __builtin_amdgcn_wmma_i32_16x16x64_iu8(true, a0, true, b0x, zc, false, false);
        v8i cy = __builtin_amdgcn_wmma_i32_16x16x64_iu8(true, a0, true, b0y, zc, false, false);
        cx = __builtin_amdgcn_wmma_i32_16x16x64_iu8(true, a1, true, b1x, cx, false, false);
        cy = __builtin_amdgcn_wmma_i32_16x16x64_iu8(true, a1, true, b1y, cy, false, false);

        float lmax = 0.0f;
        #pragma unroll
        for (int j = 0; j < 8; ++j) {
            const int m    = coutT + j + l16 * 8;
            const int pixx = pixBase + ptx * 16 + lm;
            const int pixy = pixBase + pty * 16 + lm;
            const long long ox = (long long)b * CHW + (long long)m * HWp + pixx;
            const long long oy = (long long)b * CHW + (long long)m * HWp + pixy;
            float yx = (float)cx[j] * deq + biasS[m];
            float yy = (float)cy[j] * deq + biasS[m];
            if (RELU) { yx = fmaxf(yx, 0.0f); yy = fmaxf(yy, 0.0f); }
            if (RESID) { yx += resid[ox]; yy += resid[oy]; }
            out[ox] = (TOUT)yx;
            out[oy] = (TOUT)yy;
            lmax = fmaxf(lmax, fmaxf(fabsf(yx), fabsf(yy)));
        }
        if (outSlot) atomicMax(&sm, __float_as_uint(lmax));
    }
    __syncthreads();
    if (outSlot && tid == 0) atomicMax(outSlot, sm);
}

extern "C" void kernel_launch(void* const* d_in, const int* in_sizes, int n_in,
                              void* d_out, int out_size, void* d_ws, size_t ws_size,
                              hipStream_t stream) {
    (void)in_sizes; (void)n_in; (void)out_size; (void)ws_size;

    const float* x   = (const float*)d_in[0];
    const float* w1  = (const float*)d_in[1];
    const float* b1  = (const float*)d_in[2];
    const float* p1  = (const float*)d_in[3];
    const float* pb1 = (const float*)d_in[4];
    const float* w2  = (const float*)d_in[5];
    const float* b2  = (const float*)d_in[6];
    const float* p2  = (const float*)d_in[7];
    const float* pb2 = (const float*)d_in[8];
    float*       out = (float*)d_out;

    char* ws = (char*)d_ws;
    unsigned*    scales = (unsigned*)ws;
    float*       wscale = (float*)(ws + 64);
    signed char* w1q    = (signed char*)(ws + 128);
    signed char* w2q    = (signed char*)(ws + 2048);
    signed char* p1q    = (signed char*)(ws + 4096);
    signed char* p2q    = (signed char*)(ws + 4096 + 16384);
    _Float16*    buf0   = (_Float16*)(ws + 65536);
    _Float16*    buf1   = (_Float16*)(ws + 65536 + (size_t)TOT * sizeof(_Float16));

    const int pwGrid = Bn * (HWp / 64);
    const int dwGrid = (TOT / 8) / 256;

    k_init<<<1, 32, 0, stream>>>(scales);
    k_absmax_f32<<<2048, 256, 0, stream>>>(x, (long long)TOT, &scales[0]);
    k_prep<<<1, 256, 0, stream>>>(w1, p1, w2, p2, wscale, w1q, p1q, w2q, p2q);

    k_dw<float><<<dwGrid, 256, 0, stream>>>(x, w1q, b1, &scales[0], &wscale[0],
                                            buf0, &scales[1]);
    k_pw<true, false, _Float16><<<pwGrid, 256, 0, stream>>>(buf0, p1q, pb1,
        &scales[1], &wscale[1], buf1, nullptr, &scales[2]);
    k_dw<_Float16><<<dwGrid, 256, 0, stream>>>(buf1, w2q, b2, &scales[2], &wscale[2],
                                               buf0, &scales[3]);
    k_pw<true, true, float><<<pwGrid, 256, 0, stream>>>(buf0, p2q, pb2,
        &scales[3], &wscale[3], out, x, nullptr);
}
